// CudaFastWeightSumPerformerLayer_15891378995478
// MI455X (gfx1250) — compile-verified, matrix operand order checked by reading, not run on hardware
//
#include <hip/hip_runtime.h>
#include <hip/hip_bf16.h>
#include <math.h>

#define DEV __device__ __forceinline__

typedef __bf16 bhalf;
typedef __attribute__((ext_vector_type(16))) __bf16 v16bf;
typedef __attribute__((ext_vector_type(8)))  __bf16 v8bf;
typedef __attribute__((ext_vector_type(8)))  float  v8f;

static constexpr int L    = 2048;
static constexpr int Bt   = 4;
static constexpr int H    = 8;
static constexpr int DH   = 64;
static constexpr int DM   = 512;
static constexpr int LB   = L * Bt;        // 8192
static constexpr int NQKV = H * 3 * DH;    // 1536
static constexpr int BH   = Bt * H;        // 32

DEV int laneid() { return threadIdx.x & 31; }
DEV int waveid() { return threadIdx.x >> 5; }
DEV int crow(int i) { return ((laneid() >> 4) << 3) + i; }  // C/D row for vgpr i
DEV int ccol()      { return laneid() & 15; }               // C/D col

// CDNA5 async global->LDS copy (16B per lane), tracked by ASYNCcnt.
// lds_off = byte offset of destination inside the workgroup LDS allocation
// (dynamic LDS starts at 0: no static LDS in these kernels).
DEV void async_b128(unsigned lds_off, const void* gptr) {
  asm volatile("global_load_async_to_lds_b128 %0, %1, off"
               :: "v"(lds_off), "v"(gptr)
               : "memory");
}
#define WAIT_ASYNC(n) asm volatile("s_wait_asynccnt " #n ::: "memory")

// A/B fragment loader for 16x32 (bf16) tile at base, row stride `stride` (elems,
// multiple of 8 so the two 16B halves are aligned). Layout per CDNA5 ISA 7.12.2.
DEV v16bf ldfrag(const bhalf* base, int stride) {
  int lane = laneid();
  const bhalf* p = base + (lane & 15) * stride + ((lane >> 4) << 3);
  v8bf lo = *reinterpret_cast<const v8bf*>(p);
  v8bf hi = *reinterpret_cast<const v8bf*>(p + 16);
  v16bf r;
#pragma unroll
  for (int i = 0; i < 8; ++i) { r[i] = lo[i]; r[i + 8] = hi[i]; }
  return r;
}

DEV v8f wmma_bf(v16bf a, v16bf b, v8f c) {
  return __builtin_amdgcn_wmma_f32_16x16x32_bf16(false, a, false, b, (short)0, c,
                                                 false, false);
}

// ---------------------------------------------------------------- convert f32->bf16
__global__ void k_cvt(const float* __restrict__ src, bhalf* __restrict__ dst, int n) {
  int i = blockIdx.x * blockDim.x + threadIdx.x;
  int stride = gridDim.x * blockDim.x;
  for (; i < n; i += stride) dst[i] = (bhalf)src[i];
}

// ---------------------------------------------------------------- QKV GEMM
// qkv[row=(l*B+b), n] = h[row,:] . w_qkv[n,:]  (K=512). Writes q/k/v to (B,H,L,DH),
// q,k pre-scaled by dh^-0.25. Double-buffered async global->LDS staging.
// LDS per buffer: A 128x40 (10240B) + B 64x40 (5120B) = 15360B; two buffers.
__global__ void __launch_bounds__(256) k_qkv(const bhalf* __restrict__ hbf,
                                             const bhalf* __restrict__ wq,
                                             bhalf* __restrict__ qo,
                                             bhalf* __restrict__ ko,
                                             bhalf* __restrict__ vo) {
  extern __shared__ __align__(16) char smem[];
  bhalf* base = (bhalf*)smem;
  const int rb = blockIdx.x / 24;
  const int cb = blockIdx.x % 24;
  const int t = threadIdx.x;
  const int w = waveid();
  v8f acc[4] = {};

  auto issue = [&](int ks, int p) {   // 3 async instructions per wave
    const int pb = p * 15360;
#pragma unroll
    for (int i = 0; i < 2; ++i) {     // A tile: 512 x 16B chunks
      int c = i * 256 + t;
      int row = c >> 2, q = c & 3;
      async_b128((unsigned)(pb + (row * 40 + q * 8) * 2),
                 hbf + (size_t)(rb * 128 + row) * DM + ks * 32 + q * 8);
    }
    {                                 // B tile: 256 x 16B chunks
      int row = t >> 2, q = t & 3;
      async_b128((unsigned)(pb + 10240 + (row * 40 + q * 8) * 2),
                 wq + (size_t)(cb * 64 + row) * DM + ks * 32 + q * 8);
    }
  };

  issue(0, 0);
  for (int ks = 0; ks < 16; ++ks) {
    int p = ks & 1;
    if (ks < 15) { issue(ks + 1, p ^ 1); WAIT_ASYNC(3); }  // keep next 3 in flight
    else        { WAIT_ASYNC(0); }
    __syncthreads();
    bhalf* As = base + p * 7680;
    bhalf* Bs = As + 5120;
    v16bf a = ldfrag(As + (w * 16) * 40, 40);
#pragma unroll
    for (int j = 0; j < 4; ++j)
      acc[j] = wmma_bf(a, ldfrag(Bs + (j * 16) * 40, 40), acc[j]);
    __syncthreads();
  }
  const int part = cb % 3;              // 0=q 1=k 2=v (each 64-col block is one part)
  const int head = cb / 3;
  bhalf* dst = (part == 0) ? qo : (part == 1) ? ko : vo;
  const float sc = (part < 2) ? 0.35355339059327373f : 1.0f;  // 64^-0.25
#pragma unroll
  for (int j = 0; j < 4; ++j)
#pragma unroll
    for (int i = 0; i < 8; ++i) {
      int row = rb * 128 + w * 16 + crow(i);
      int d = j * 16 + ccol();
      int l = row >> 2, b = row & 3;
      dst[((size_t)(b * H + head) * L + l) * DH + d] = (bhalf)(acc[j][i] * sc);
    }
}

// ---------------------------------------------------------------- prime features
// For 32 rows of one head: d = Xn @ proj (WMMA), feat = [exp(d-diag), exp(-d-diag)],
// row-normalized, stored bf16. Deterministic LDS tree reductions (no float atomics).
__global__ void __launch_bounds__(256) k_prime(const bhalf* __restrict__ xin,
                                               const bhalf* __restrict__ projbf,
                                               bhalf* __restrict__ fpout) {
  extern __shared__ __align__(16) char smem[];
  bhalf* PT = (bhalf*)smem;              // 256 x 72  (projT[n][k])
  bhalf* Xn = PT + 256 * 72;             // 32 x 72
  float* E = (float*)(Xn + 32 * 72);     // 32 x 512
  float* diag = E + 32 * 512;            // 32
  float* rinv = diag + 32;               // 32
  float* red = rinv + 32;                // 32 x 8

  const int bh = blockIdx.x >> 6;
  const int l0 = (blockIdx.x & 63) * 32;
  const int t = threadIdx.x;
  const int w = waveid();

#pragma unroll 8
  for (int i = 0; i < 64; ++i) {         // proj (64x256) -> PT[n][k]
    int flat = i * 256 + t;
    int k = flat >> 8, n = flat & 255;
    PT[n * 72 + k] = projbf[k * 256 + n];
  }
#pragma unroll 4
  for (int i = 0; i < 8; ++i) {
    int flat = i * 256 + t;
    int row = flat >> 6, k = flat & 63;
    Xn[row * 72 + k] = xin[((size_t)bh * L + l0 + row) * DH + k];
  }
  __syncthreads();
  {                                       // diag = 0.5*||xn||^2
    int row = t >> 3, part = t & 7;
    float s = 0.f;
#pragma unroll
    for (int e = 0; e < 8; ++e) { float v = (float)Xn[row * 72 + part * 8 + e]; s += v * v; }
    red[row * 8 + part] = s;
  }
  __syncthreads();
  if (t < 32) {
    float s = 0.f;
#pragma unroll
    for (int p = 0; p < 8; ++p) s += red[t * 8 + p];
    diag[t] = 0.5f * s;
  }
  __syncthreads();
  const int mt = w & 1;
  v8f acc[4] = {};
#pragma unroll
  for (int ks = 0; ks < 2; ++ks) {
    v16bf a = ldfrag(Xn + mt * 16 * 72 + ks * 32, 72);
#pragma unroll
    for (int j = 0; j < 4; ++j) {
      int nt = (w >> 1) * 4 + j;
      acc[j] = wmma_bf(a, ldfrag(PT + nt * 16 * 72 + ks * 32, 72), acc[j]);
    }
  }
#pragma unroll
  for (int j = 0; j < 4; ++j) {
    int nt = (w >> 1) * 4 + j;
#pragma unroll
    for (int i = 0; i < 8; ++i) {
      int row = mt * 16 + crow(i);
      int n = nt * 16 + ccol();
      float dg = diag[row];
      E[row * 512 + n] = __expf(acc[j][i] - dg);
      E[row * 512 + 256 + n] = __expf(-acc[j][i] - dg);
    }
  }
  __syncthreads();
  {                                       // deterministic row sums over 512 feats
    int row = t >> 3, part = t & 7;
    float s = 0.f;
#pragma unroll 8
    for (int e = 0; e < 64; ++e) s += E[row * 512 + part * 64 + e];
    red[row * 8 + part] = s;
  }
  __syncthreads();
  if (t < 32) {
    float s = 0.f;
#pragma unroll
    for (int p = 0; p < 8; ++p) s += red[t * 8 + p];
    rinv[t] = 1.0f / s;
  }
  __syncthreads();
#pragma unroll 8
  for (int i = 0; i < 64; ++i) {
    int flat = i * 256 + t;
    int row = flat >> 9, f = flat & 511;
    fpout[((size_t)bh * L + l0 + row) * 512 + f] = (bhalf)(E[row * 512 + f] * rinv[row]);
  }
}

// ---------------------------------------------------------------- chunked causal scan
// One WG per (b,h). Chunk C=32. State W^T (64x512 bf16) + r (512 f32) in LDS.
// Qc for chunk t+1 is async-prefetched during the W update phase of chunk t.
__global__ void __launch_bounds__(256) k_scan(const bhalf* __restrict__ qp,
                                              const bhalf* __restrict__ kp,
                                              const bhalf* __restrict__ vb,
                                              float* __restrict__ outp,
                                              float* __restrict__ den) {
  extern __shared__ __align__(16) char smem[];
  bhalf* WT = (bhalf*)smem;              // 64 x 520  (W^T[d][f])
  bhalf* Qc = WT + 64 * 520;             // 32 x 520
  bhalf* Kc = Qc + 32 * 520;             // 32 x 520
  bhalf* KT = Kc + 32 * 520;             // 512 x 40  (K^T[f][l])
  bhalf* VT = KT + 512 * 40;             // 64 x 40   (V^T[d][l])
  bhalf* Sb = VT + 64 * 40;              // 32 x 40   masked S, bf16
  float* Sf = (float*)(Sb + 32 * 40);    // 32 x 32
  float* rs = Sf + 32 * 32;              // 512
  float* red = rs + 512;                 // 32 x 8
  constexpr unsigned QC_OFF = 64u * 520u * 2u;   // byte offset of Qc in LDS

  const int bh = blockIdx.x;
  const int t = threadIdx.x;
  const int w = waveid();
  const bhalf* qpB = qp + (size_t)bh * L * 512;
  const bhalf* kpB = kp + (size_t)bh * L * 512;
  const bhalf* vB = vb + (size_t)bh * L * 64;
  float* outB = outp + (size_t)bh * L * 64;
  float* denB = den + (size_t)bh * L;

  auto issueQ = [&](int tq) {            // 8 async instructions per wave
#pragma unroll
    for (int i = 0; i < 8; ++i) {
      int c = i * 256 + t;               // 2048 x 16B chunks (32 rows x 64 chunks)
      int row = c >> 6, q = c & 63;
      async_b128(QC_OFF + (unsigned)(row * 520 + q * 8) * 2u,
                 qpB + (size_t)(tq + row) * 512 + q * 8);
    }
  };

  issueQ(0);
  for (int i = t; i < 64 * 520; i += 256) WT[i] = (bhalf)0.f;
  for (int i = t; i < 512; i += 256) rs[i] = 0.f;
  __syncthreads();

  for (int t0 = 0; t0 < L; t0 += 32) {
    // 1. stage Kc/KT/VT via VGPRs (KT needs the transpose scatter); Qc is async.
#pragma unroll 8
    for (int i = 0; i < 64; ++i) {
      int flat = i * 256 + t;
      int row = flat >> 9, f = flat & 511;
      bhalf kv = kpB[(size_t)(t0 + row) * 512 + f];
      Kc[row * 520 + f] = kv;
      KT[f * 40 + row] = kv;
    }
#pragma unroll 4
    for (int i = 0; i < 8; ++i) {
      int flat = i * 256 + t;
      int row = flat >> 6, d = flat & 63;
      VT[d * 40 + row] = vB[(size_t)(t0 + row) * 64 + d];
    }
    WAIT_ASYNC(0);                       // Qc chunk complete
    __syncthreads();
    // 2. S = Qc @ Kc^T   (waves 0..3, K=512)
    if (w < 4) {
      int mt = w >> 1, nt = w & 1;
      v8f s = {};
#pragma unroll
      for (int ks = 0; ks < 16; ++ks)
        s = wmma_bf(ldfrag(Qc + mt * 16 * 520 + ks * 32, 520),
                    ldfrag(Kc + nt * 16 * 520 + ks * 32, 520), s);
#pragma unroll
      for (int i = 0; i < 8; ++i)
        Sf[(mt * 16 + crow(i)) * 32 + nt * 16 + ccol()] = s[i];
    }
    __syncthreads();
    // 3. causal mask (inclusive) -> bf16
#pragma unroll
    for (int i = 0; i < 4; ++i) {
      int flat = i * 256 + t;
      int iq = flat >> 5, jk = flat & 31;
      Sb[iq * 40 + jk] = (bhalf)((jk <= iq) ? Sf[iq * 32 + jk] : 0.f);
    }
    __syncthreads();
    // 4. out = Qc @ W (inter, K=512) + Sb @ V (intra, K=32); one 16x16 tile per wave
    {
      int mt = w >> 2, nt = w & 3;
      v8f o = {};
#pragma unroll
      for (int ks = 0; ks < 16; ++ks)
        o = wmma_bf(ldfrag(Qc + mt * 16 * 520 + ks * 32, 520),
                    ldfrag(WT + nt * 16 * 520 + ks * 32, 520), o);
      o = wmma_bf(ldfrag(Sb + mt * 16 * 40, 40), ldfrag(VT + nt * 16 * 40, 40), o);
#pragma unroll
      for (int i = 0; i < 8; ++i)
        outB[(size_t)(t0 + mt * 16 + crow(i)) * 64 + nt * 16 + ccol()] = o[i];
    }
    __syncthreads();
    // 5. denom_t = q_t . r_prev + prefix-rowsum(S); then r += colsum over chunk of kp
    {
      int row = t >> 3, part = t & 7;
      float s = 0.f;
#pragma unroll 8
      for (int e = 0; e < 64; ++e) {
        int f = part * 64 + e;
        s += (float)Qc[row * 520 + f] * rs[f];
      }
      red[row * 8 + part] = s;
    }
    __syncthreads();
    if (t < 32) {
      float s = 0.f;
#pragma unroll
      for (int p = 0; p < 8; ++p) s += red[t * 8 + p];
      for (int j = 0; j <= t; ++j) s += Sf[t * 32 + j];
      denB[t0 + t] = s;
    }
    __syncthreads();
#pragma unroll
    for (int rep = 0; rep < 2; ++rep) {
      int ff = t + rep * 256;
      float s = 0.f;
#pragma unroll 8
      for (int row = 0; row < 32; ++row) s += (float)Kc[row * 520 + ff];
      rs[ff] += s;
    }
    __syncthreads();
    // Qc is dead from here: prefetch next chunk's Qc under the W-update WMMAs.
    if (t0 + 32 < L) issueQ(t0 + 32);
    // 6. W^T += V^T @ Kc  (128 tiles of 16x16, K=32; 16 per wave)
#pragma unroll 2
    for (int idx = 0; idx < 16; ++idx) {
      int tile = w * 16 + idx;
      int dt = tile >> 5, ft = tile & 31;
      v8f dv = {};
      dv = wmma_bf(ldfrag(VT + dt * 16 * 40, 40), ldfrag(KT + ft * 16 * 40, 40), dv);
#pragma unroll
      for (int i = 0; i < 8; ++i) {
        int di = (dt * 16 + crow(i)) * 520 + ft * 16 + ccol();
        WT[di] = (bhalf)((float)WT[di] + dv[i]);
      }
    }
    __syncthreads();
  }
}

// ---------------------------------------------------------------- out-proj + residual + LN
// Double-buffered async staging of the 512x32 w_o slab.
__global__ void __launch_bounds__(256) k_outln(const float* __restrict__ outp,
                                               const float* __restrict__ den,
                                               const bhalf* __restrict__ wob,
                                               const float* __restrict__ hin,
                                               const float* __restrict__ gamma,
                                               const float* __restrict__ beta,
                                               float* __restrict__ yout) {
  extern __shared__ __align__(16) char smem[];
  bhalf* Aa = (bhalf*)smem;                       // 16 x 520        (16640 B)
  bhalf* Bs0 = Aa + 16 * 520;                     // 512 x 40 x 2    (81920 B)
  float* X = (float*)(Bs0 + 2 * 512 * 40);        // 16 x 512
  float* red = X + 16 * 512;                      // 16 x 16
  float* mu = red + 16 * 16;                      // 16
  float* rstd = mu + 16;                          // 16
  constexpr unsigned BS_OFF = 16u * 520u * 2u;    // 16640

  const int r0 = blockIdx.x * 16;
  const int t = threadIdx.x;
  const int w = waveid();

  auto issueB = [&](int ks, int p) {              // 8 async instructions per wave
    const unsigned pb = BS_OFF + (unsigned)p * 40960u;
#pragma unroll
    for (int i = 0; i < 8; ++i) {
      int c = i * 256 + t;                        // 2048 x 16B chunks
      int n = c >> 2, q = c & 3;
      async_b128(pb + (unsigned)(n * 40 + q * 8) * 2u,
                 wob + (size_t)n * 512 + ks * 32 + q * 8);
    }
  };

  issueB(0, 0);
#pragma unroll 4
  for (int i = 0; i < 32; ++i) {
    int flat = i * 256 + t;
    int row = flat >> 9, col = flat & 511;
    int head = col >> 6, d = col & 63;
    int grow = r0 + row;
    int l = grow >> 2, b = grow & 3;
    int bh = b * H + head;
    float o = outp[((size_t)bh * L + l) * 64 + d];
    float dn = den[(size_t)bh * L + l] + 1e-5f;
    Aa[row * 520 + col] = (bhalf)(0.125f * o / dn);      // scale = dh^-0.5
    X[row * 512 + col] = hin[(size_t)grow * DM + col];   // residual
  }
  __syncthreads();
  v8f acc[4] = {};
  for (int ks = 0; ks < 16; ++ks) {
    int p = ks & 1;
    if (ks < 15) { issueB(ks + 1, p ^ 1); WAIT_ASYNC(8); }
    else        { WAIT_ASYNC(0); }
    __syncthreads();
    bhalf* Bs = Bs0 + p * 512 * 40;
    v16bf a = ldfrag(Aa + ks * 32, 520);
#pragma unroll
    for (int j = 0; j < 4; ++j)
      acc[j] = wmma_bf(a, ldfrag(Bs + (w * 64 + j * 16) * 40, 40), acc[j]);
    __syncthreads();
  }
#pragma unroll
  for (int j = 0; j < 4; ++j)
#pragma unroll
    for (int i = 0; i < 8; ++i) {
      int row = crow(i), n = w * 64 + j * 16 + ccol();
      X[row * 512 + n] += acc[j][i];
    }
  __syncthreads();
  {
    int row = t >> 4, part = t & 15;
    float s = 0.f;
#pragma unroll 8
    for (int e = 0; e < 32; ++e) s += X[row * 512 + part * 32 + e];
    red[row * 16 + part] = s;
  }
  __syncthreads();
  if (t < 16) {
    float s = 0.f;
#pragma unroll
    for (int p = 0; p < 16; ++p) s += red[t * 16 + p];
    mu[t] = s * (1.0f / 512.0f);
  }
  __syncthreads();
  {
    int row = t >> 4, part = t & 15;
    float m = mu[row], s = 0.f;
#pragma unroll 8
    for (int e = 0; e < 32; ++e) {
      float dx = X[row * 512 + part * 32 + e] - m;
      s += dx * dx;
    }
    red[row * 16 + part] = s;
  }
  __syncthreads();
  if (t < 16) {
    float s = 0.f;
#pragma unroll
    for (int p = 0; p < 16; ++p) s += red[t * 16 + p];
    rstd[t] = rsqrtf(s * (1.0f / 512.0f) + 1e-5f);
  }
  __syncthreads();
#pragma unroll 4
  for (int i = 0; i < 32; ++i) {
    int flat = i * 256 + t;
    int row = flat >> 9, col = flat & 511;
    float v = (X[row * 512 + col] - mu[row]) * rstd[row];
    yout[(size_t)(r0 + row) * DM + col] = v * gamma[col] + beta[col];
  }
}

// ----------------------------------------------------------------------------
extern "C" void kernel_launch(void* const* d_in, const int* in_sizes, int n_in,
                              void* d_out, int out_size, void* d_ws, size_t ws_size,
                              hipStream_t stream) {
  (void)in_sizes; (void)n_in; (void)out_size; (void)ws_size;
  const float* h     = (const float*)d_in[0];
  const float* w_qkv = (const float*)d_in[1];
  const float* w_o   = (const float*)d_in[2];
  const float* gamma = (const float*)d_in[3];
  const float* beta  = (const float*)d_in[4];
  const float* proj  = (const float*)d_in[5];

  char* ws = (char*)d_ws;
  size_t off = 0;
  auto alloc = [&](size_t bytes) {
    char* p = ws + off;
    off += (bytes + 255) & ~(size_t)255;
    return p;
  };
  bhalf* h_bf  = (bhalf*)alloc((size_t)LB * DM * 2);
  bhalf* wq_bf = (bhalf*)alloc((size_t)NQKV * DM * 2);
  bhalf* wo_bf = (bhalf*)alloc((size_t)DM * DM * 2);
  bhalf* pj_bf = (bhalf*)alloc((size_t)DH * 256 * 2);
  bhalf* q_bf  = (bhalf*)alloc((size_t)BH * L * DH * 2);
  bhalf* k_bf  = (bhalf*)alloc((size_t)BH * L * DH * 2);
  bhalf* v_bf  = (bhalf*)alloc((size_t)BH * L * DH * 2);
  bhalf* qp_bf = (bhalf*)alloc((size_t)BH * L * 512 * 2);
  bhalf* kp_bf = (bhalf*)alloc((size_t)BH * L * 512 * 2);
  float* outf  = (float*)alloc((size_t)BH * L * DH * 4);
  float* denf  = (float*)alloc((size_t)BH * L * 4);

  const size_t shA = (size_t)2 * (128 * 40 + 64 * 40) * 2;         // 30720
  const size_t shP = (size_t)(256 * 72 + 32 * 72) * 2 +
                     (size_t)(32 * 512 + 32 + 32 + 32 * 8) * 4;
  const size_t shS = (size_t)(64 * 520 + 32 * 520 + 32 * 520 + 512 * 40 + 64 * 40 + 32 * 40) * 2 +
                     (size_t)(32 * 32 + 512 + 32 * 8) * 4;
  const size_t shD = (size_t)(16 * 520 + 2 * 512 * 40) * 2 +
                     (size_t)(16 * 512 + 16 * 16 + 16 + 16) * 4;   // 132480

  (void)hipFuncSetAttribute((const void*)k_prime,
                            hipFuncAttributeMaxDynamicSharedMemorySize, (int)shP);
  (void)hipFuncSetAttribute((const void*)k_scan,
                            hipFuncAttributeMaxDynamicSharedMemorySize, (int)shS);
  (void)hipFuncSetAttribute((const void*)k_outln,
                            hipFuncAttributeMaxDynamicSharedMemorySize, (int)shD);

  k_cvt<<<2048, 256, 0, stream>>>(h, h_bf, LB * DM);
  k_cvt<<<512, 256, 0, stream>>>(w_qkv, wq_bf, NQKV * DM);
  k_cvt<<<256, 256, 0, stream>>>(w_o, wo_bf, DM * DM);
  k_cvt<<<16, 256, 0, stream>>>(proj, pj_bf, DH * 256);

  k_qkv<<<64 * 24, 256, shA, stream>>>(h_bf, wq_bf, q_bf, k_bf, v_bf);
  k_prime<<<BH * (L / 32), 256, shP, stream>>>(q_bf, pj_bf, qp_bf);
  k_prime<<<BH * (L / 32), 256, shP, stream>>>(k_bf, pj_bf, kp_bf);
  k_scan<<<BH, 256, shS, stream>>>(qp_bf, kp_bf, v_bf, outf, denf);
  k_outln<<<LB / 16, 256, shD, stream>>>(outf, denf, wo_bf, h, gamma, beta,
                                         (float*)d_out);
}
